// SOMLayer_71193377898768
// MI455X (gfx1250) — matrix operand order repaired and checked by reading, not compiled, not run on hardware
//
#include <hip/hip_runtime.h>
#include <hip/hip_bf16.h>
#include <stdint.h>

typedef __bf16 bf16_t;
typedef __attribute__((ext_vector_type(16))) __bf16 v16bf;
typedef __attribute__((ext_vector_type(8)))  float  v8f;

#define B_DIM   32
#define T_DIM   1024
#define D_DIM   512
#define N_NODES 1024
#define M_ROWS  (B_DIM * T_DIM)
#define MAXSEQ  4000
#define BLENDF  0.1f

#define MTILE   64      // rows per workgroup
#define NWAVES  16      // 512 threads

union AB16 { uint4 u[2]; v16bf v; };

// ---------------------------------------------------------------------------
// Kernel 1: nodes f32 -> bf16 + per-node squared norm
// ---------------------------------------------------------------------------
__global__ __launch_bounds__(256) void prep_nodes_kernel(
    const float* __restrict__ nodes, bf16_t* __restrict__ nodesb,
    float* __restrict__ n_sq) {
  __shared__ float ssum;
  const int n = blockIdx.x;
  const int tid = threadIdx.x;
  if (tid == 0) ssum = 0.0f;
  __syncthreads();
  const float2 v = ((const float2*)(nodes + (size_t)n * D_DIM))[tid];
  float p = v.x * v.x + v.y * v.y;
  nodesb[(size_t)n * D_DIM + tid * 2 + 0] = (bf16_t)v.x;
  nodesb[(size_t)n * D_DIM + tid * 2 + 1] = (bf16_t)v.y;
  #pragma unroll
  for (int off = 16; off > 0; off >>= 1) p += __shfl_xor(p, off, 32);
  if ((tid & 31) == 0) atomicAdd(&ssum, p);
  __syncthreads();
  if (tid == 0) n_sq[n] = ssum;
}

// ---------------------------------------------------------------------------
// Kernel 2: fused GEMM (bf16 WMMA) + distance/q/row-sum/argmin epilogue.
// One workgroup = 64 rows x all 1024 nodes. 16 waves; each wave owns a
// 64x64 strip: 4 m-frags x 4 n-frags. B loaded once per (k, nt), reused by
// 4 WMMAs -> 2 wmma per global_load_b128 (L2 traffic for nodes: 512 MB total).
// ---------------------------------------------------------------------------
__global__ __launch_bounds__(512) void som_main_kernel(
    const float* __restrict__ z, const float* __restrict__ tw,
    const bf16_t* __restrict__ nodesb, const float* __restrict__ n_sq,
    float* __restrict__ q_out, float* __restrict__ bmu_f,
    int* __restrict__ bmu_i) {
  __shared__ __align__(16) bf16_t As[MTILE][528];   // 64 rows, padded stride
  __shared__ float zsq_s[MTILE];
  __shared__ float red_sum[NWAVES][MTILE];
  __shared__ float red_min[NWAVES][MTILE];
  __shared__ int   red_idx[NWAVES][MTILE];
  __shared__ float inv_s[MTILE];

  const int tid = threadIdx.x;
  const int m0  = blockIdx.x * MTILE;

  if (tid < MTILE) zsq_s[tid] = 0.0f;
  __syncthreads();

  // Stage A: 64 rows of (z * time_weight), f32 -> bf16 into LDS, + row |.|^2
  {
    const int row = tid >> 3;          // 0..63
    const int seg = tid & 7;           // 8 segments x 64 elems
    const int m = m0 + row;
    const int t = m & (T_DIM - 1);
    const float w = tw[(MAXSEQ - T_DIM) + t];
    const float4* src = (const float4*)(z + (size_t)m * D_DIM + seg * 64);
    float acc = 0.0f;
    #pragma unroll
    for (int i = 0; i < 16; ++i) {
      float4 v = src[i];
      v.x *= w; v.y *= w; v.z *= w; v.w *= w;
      acc += v.x * v.x + v.y * v.y + v.z * v.z + v.w * v.w;
      const int c = seg * 64 + i * 4;
      As[row][c + 0] = (bf16_t)v.x;
      As[row][c + 1] = (bf16_t)v.y;
      As[row][c + 2] = (bf16_t)v.z;
      As[row][c + 3] = (bf16_t)v.w;
    }
    atomicAdd(&zsq_s[row], acc);
  }
  __syncthreads();

  const int wv    = tid >> 5;      // wave 0..15
  const int lane  = tid & 31;
  const int hh    = lane >> 4;     // lane half
  const int lr    = lane & 15;
  const int nbase = wv * 64;       // each wave owns 64 node columns

  v8f acc[4][4];
  const v8f vzero = {0.f, 0.f, 0.f, 0.f, 0.f, 0.f, 0.f, 0.f};
  #pragma unroll
  for (int mt = 0; mt < 4; ++mt)
    #pragma unroll
    for (int nt = 0; nt < 4; ++nt) acc[mt][nt] = vzero;

  // K loop: D=512 in steps of 32 (v_wmma_f32_16x16x32_bf16)
  for (int k0 = 0; k0 < D_DIM; k0 += 32) {
    AB16 af[4];   // A frags for the 4 m-tiles (lanes 0-15 K+0, 16-31 K+8)
    #pragma unroll
    for (int mt = 0; mt < 4; ++mt) {
      af[mt].u[0] = *(const uint4*)(&As[mt * 16 + lr][k0 + hh * 8]);
      af[mt].u[1] = *(const uint4*)(&As[mt * 16 + lr][k0 + 16 + hh * 8]);
    }
    #pragma unroll
    for (int nt = 0; nt < 4; ++nt) {
      const int n = nbase + nt * 16 + lr;
      AB16 bfm;  // B fragment: col n, K = k0 + hh*8 .. +7, k0 + 16 + hh*8 .. +7
      const uint4* bp = (const uint4*)(nodesb + (size_t)n * D_DIM + k0 + hh * 8);
      bfm.u[0] = bp[0];
      bfm.u[1] = bp[2];
      #pragma unroll
      for (int mt = 0; mt < 4; ++mt)
        acc[mt][nt] = __builtin_amdgcn_wmma_f32_16x16x32_bf16(
            false, af[mt].v, false, bfm.v, (short)0, acc[mt][nt], false, false);
    }
  }

  // Epilogue: d2 -> dist -> q, per-lane partial row-sum + argmin, per m-tile
  #pragma unroll
  for (int mt = 0; mt < 4; ++mt) {
    float rs[8], mind[8];
    int mini[8];
    const float zqv[1] = {0};
    (void)zqv;
    float zq[8];
    #pragma unroll
    for (int r = 0; r < 8; ++r) {
      zq[r] = zsq_s[mt * 16 + r + 8 * hh];
      rs[r] = 0.f; mind[r] = 3.402823466e38f; mini[r] = 0;
    }
    #pragma unroll
    for (int nt = 0; nt < 4; ++nt) {
      const int n = nbase + nt * 16 + lr;
      const float ns = n_sq[n];
      #pragma unroll
      for (int r = 0; r < 8; ++r) {
        float d2 = zq[r] + ns - 2.0f * acc[mt][nt][r];
        d2 = fmaxf(d2, 0.0f);
        const float dist = sqrtf(d2);
        const float qv = 1.0f / (1.0f + dist);
        acc[mt][nt][r] = qv;     // keep q in accum regs until normalized
        rs[r] += qv;
        if (dist < mind[r]) { mind[r] = dist; mini[r] = n; }
      }
    }
    // Reduce across the 16 lanes that share each row
    #pragma unroll
    for (int off = 8; off > 0; off >>= 1) {
      #pragma unroll
      for (int r = 0; r < 8; ++r) {
        rs[r] += __shfl_xor(rs[r], off, 16);
        const float od = __shfl_xor(mind[r], off, 16);
        const int   oi = __shfl_xor(mini[r], off, 16);
        if (od < mind[r] || (od == mind[r] && oi < mini[r])) { mind[r] = od; mini[r] = oi; }
      }
    }
    if (lr < 8) {
      const int rowi = mt * 16 + lr + 8 * hh;
      red_sum[wv][rowi] = rs[lr];
      red_min[wv][rowi] = mind[lr];
      red_idx[wv][rowi] = mini[lr];
    }
  }
  __syncthreads();

  // Cross-wave reduce: final 1/rowsum + BMU per row
  if (tid < MTILE) {
    float s = 0.0f, md = 3.402823466e38f;
    int mi = 0x7fffffff;
    #pragma unroll
    for (int w2 = 0; w2 < NWAVES; ++w2) {
      s += red_sum[w2][tid];
      const float d = red_min[w2][tid];
      const int   i = red_idx[w2][tid];
      if (d < md || (d == md && i < mi)) { md = d; mi = i; }
    }
    inv_s[tid] = 1.0f / s;
    bmu_i[m0 + tid] = mi;
    bmu_f[m0 + tid] = (float)mi;
  }
  __syncthreads();

  // Normalized q store (lanes 0-15 / 16-31 each cover 16 contiguous cols)
  #pragma unroll
  for (int mt = 0; mt < 4; ++mt) {
    float inv[8];
    #pragma unroll
    for (int r = 0; r < 8; ++r) inv[r] = inv_s[mt * 16 + r + 8 * hh];
    #pragma unroll
    for (int nt = 0; nt < 4; ++nt) {
      const int n = nbase + nt * 16 + lr;
      #pragma unroll
      for (int r = 0; r < 8; ++r) {
        const int m = m0 + mt * 16 + r + 8 * hh;
        q_out[(size_t)m * N_NODES + n] = acc[mt][nt][r] * inv[r];
      }
    }
  }
}

// ---------------------------------------------------------------------------
// Kernel 3: som_z = z + 0.1*(nodes[bmu] - z)  (pure bandwidth)
// ---------------------------------------------------------------------------
__global__ __launch_bounds__(256) void som_blend_kernel(
    const float* __restrict__ z, const float* __restrict__ nodes,
    const int* __restrict__ bmu, float* __restrict__ som_z) {
  const int m = blockIdx.x * 2 + (threadIdx.x >> 7);
  const int c = (threadIdx.x & 127) * 4;
  const int b = bmu[m];
  const float4 zv = *(const float4*)(z + (size_t)m * D_DIM + c);
  const float4 nv = *(const float4*)(nodes + (size_t)b * D_DIM + c);
  float4 o;
  o.x = zv.x + BLENDF * (nv.x - zv.x);
  o.y = zv.y + BLENDF * (nv.y - zv.y);
  o.z = zv.z + BLENDF * (nv.z - zv.z);
  o.w = zv.w + BLENDF * (nv.w - zv.w);
  *(float4*)(som_z + (size_t)m * D_DIM + c) = o;
}

// ---------------------------------------------------------------------------
extern "C" void kernel_launch(void* const* d_in, const int* in_sizes, int n_in,
                              void* d_out, int out_size, void* d_ws, size_t ws_size,
                              hipStream_t stream) {
  const float* z     = (const float*)d_in[0];
  const float* nodes = (const float*)d_in[1];
  const float* tw    = (const float*)d_in[2];

  float* out   = (float*)d_out;
  float* som_z = out;                                                 // B*T*D
  float* q_out = out + (size_t)M_ROWS * D_DIM;                        // M*N
  float* bmu_f = out + (size_t)M_ROWS * D_DIM + (size_t)M_ROWS * N_NODES;

  char* ws = (char*)d_ws;
  bf16_t* nodesb = (bf16_t*)ws;                                       // 1 MB
  float*  n_sq   = (float*)(ws + (size_t)N_NODES * D_DIM * 2);        // 4 KB
  int*    bmu_i  = (int*)(ws + (size_t)N_NODES * D_DIM * 2 + (size_t)N_NODES * 4);

  prep_nodes_kernel<<<N_NODES, 256, 0, stream>>>(nodes, nodesb, n_sq);
  som_main_kernel<<<M_ROWS / MTILE, 512, 0, stream>>>(z, tw, nodesb, n_sq,
                                                      q_out, bmu_f, bmu_i);
  som_blend_kernel<<<M_ROWS / 2, 256, 0, stream>>>(z, nodes, bmu_i, som_z);
}
